// MultimodalFusion_35287451304351
// MI455X (gfx1250) — matrix operand order repaired and checked by reading, not compiled, not run on hardware
//
#include <hip/hip_runtime.h>
#include <hip/hip_bf16.h>
#include <math.h>

// ---------------------------------------------------------------------------
// Multimodal fusion block for MI455X (gfx1250, wave32, WMMA + TDM).
// Heavy GEMMs: v_wmma_f32_16x16x32_f16 (f16 in, f32 accum); weight panels are
// staged into LDS by the Tensor Data Mover (tensor_load_to_lds + TENSORcnt),
// with TDM row padding chosen for bank-conflict-free ds_load_b128 reads.
// ---------------------------------------------------------------------------

typedef __attribute__((ext_vector_type(16))) _Float16 v16h;
typedef __attribute__((ext_vector_type(8)))  _Float16 v8h;
typedef __attribute__((ext_vector_type(2)))  _Float16 v2h;
typedef __attribute__((ext_vector_type(8)))  float    v8f;
typedef __attribute__((ext_vector_type(4)))  unsigned int u32x4;
typedef __attribute__((ext_vector_type(8)))  int      i32x8;
typedef __attribute__((ext_vector_type(4)))  int      i32x4;

#define MFEAT   512
#define NHEADS  8
#define DHEAD   64
#define FFNDIM  2048
#define NMODS   4
#define NTOK    16384               // 64 * 256
#define NROWS   (NTOK * NMODS)      // 65536

#define KCHUNK  256                 // K staged per TDM load (halves)
#define BROWS   64                  // weight rows (output cols) per block
#define LDS_STRIDE 264              // 256 + 8 halves (TDM pad: +16B per 512B row)

// ------------------------------ WMMA fragment loads ------------------------
// A: 16x32 f16, row-major source. Contiguous k layout (B uses the same): lane = m + 16*half,
//   VGPR v holds K pair { 2v + 16*half , +1 }.
//   Byte-contiguous -> compiler emits two global_load_b128 per fragment.
__device__ __forceinline__ v16h load_fragA(const _Float16* __restrict__ A,
                                           size_t lda, int row0, int k0, int lane) {
  const int m = lane & 15;
  const int half = lane >> 4;
  const _Float16* p = A + (size_t)(row0 + m) * lda + (size_t)k0 + 16 * half;
  v16h a;
#pragma unroll
  for (int v = 0; v < 8; ++v) {
    const int koff = 2 * v;
    v2h t = *(const v2h*)(p + koff);
    a[2 * v]     = t[0];
    a[2 * v + 1] = t[1];
  }
  return a;
}

// B fragment from the LDS-staged weight panel. B(k,n) = W[n][k]; a lane's 16
// K-values are contiguous -> two ds_load_b128. Row stride 264 halves (528B =
// 132 DWORDs == 4 banks mod 64) -> the 16 lanes cover all 64 LDS banks.
__device__ __forceinline__ v16h load_fragB_lds(const _Float16* Bs, int nrow0,
                                               int k0, int lane) {
  const int n = lane & 15;
  const int half = lane >> 4;
  const _Float16* p = Bs + (size_t)(nrow0 + n) * LDS_STRIDE + k0 + 16 * half;
  v8h lo = *(const v8h*)(p);
  v8h hi = *(const v8h*)(p + 8);
  v16h b;
#pragma unroll
  for (int j = 0; j < 8; ++j) { b[j] = lo[j]; b[8 + j] = hi[j]; }
  return b;
}

// C/D: VGPR j -> row = row0 + j + 8*half, col = n (lane&15).
// EPI: 0 = +bias -> f32 ; 1 = +bias -> f16 ; 2 = +bias, exact GELU -> f16
template <int EPI>
__device__ __forceinline__ void store_tile(v8f c, float* __restrict__ Cf,
                                           _Float16* __restrict__ Ch,
                                           const float* __restrict__ bias,
                                           size_t ldc, int row0, int col, int lane) {
  const int half = lane >> 4;
  const float bv = bias[col];
#pragma unroll
  for (int j = 0; j < 8; ++j) {
    const int row = row0 + j + 8 * half;
    float v = c[j] + bv;
    if (EPI == 2) v = 0.5f * v * (1.0f + erff(v * 0.70710678118654752f));
    if (EPI == 0) Cf[(size_t)row * ldc + col] = v;
    else          Ch[(size_t)row * ldc + col] = (_Float16)v;
  }
}

// Issue one TDM load of a (KCHUNK x BROWS) f16 weight tile into LDS.
// D# built per CDNA5 ISA ch.8: group0 = {count, lds_addr, global_addr, type},
// group1 = {flags/pad, tensor dims, tile dims, dim0 stride}.
__device__ __forceinline__ void tdm_load_weight_tile(const _Float16* W, int ldw,
                                                     int nrow_base, int kc,
                                                     _Float16* Bs) {
  const unsigned lds_off = (unsigned)(size_t)(&Bs[0]);   // flat addr low 32b = LDS offset
  const unsigned long long ga =
      (unsigned long long)(size_t)(W + (size_t)nrow_base * ldw + kc);
  u32x4 g0;
  g0[0] = 1u;                                            // count=1, user mode
  g0[1] = lds_off;                                       // lds_addr
  g0[2] = (unsigned)(ga & 0xffffffffull);                // global_addr[31:0]
  g0[3] = (unsigned)((ga >> 32) & 0x1ffffffull) | (2u << 30);  // addr[56:32] | type=2
  i32x8 g1;
  // data_size=1 (2B) | pad_enable | pad_interval=6 (512B) | pad_amount=3 (16B)
  g1[0] = 0x07910000;
  g1[1] = (int)((unsigned)KCHUNK << 16);                 // tensor_dim0 = KCHUNK (lo16)
  g1[2] = (int)((unsigned)BROWS << 16);                  // dim0 hi16=0 | tensor_dim1=BROWS
  g1[3] = (int)((unsigned)KCHUNK << 16);                 // dim1 hi16=0 | tile_dim0=KCHUNK
  g1[4] = BROWS;                                         // tile_dim1=BROWS, tile_dim2=0
  g1[5] = ldw;                                           // tensor_dim0_stride[31:0]
  g1[6] = 0;
  g1[7] = 0;
  const i32x4 gz  = {0, 0, 0, 0};
  const i32x8 gz8 = {0, 0, 0, 0, 0, 0, 0, 0};
  __builtin_amdgcn_tensor_load_to_lds(g0, g1, gz, gz, gz8, 0);
}

// 4 waves / block stacked in M; block tile 128 rows x 64 cols. Each wave owns
// a 32x64 output tile: 2 A-frags (global) x 4 B-frags (LDS) -> 8 WMMA/K-step.
// Weight panel staged in KCHUNK slices by the TDM. C[i,o] = sum_k A[i,k]*W[o,k] + bias[o].
template <int EPI>
__global__ __launch_bounds__(128)
void gemm_wmma_kernel(const _Float16* __restrict__ A, const _Float16* __restrict__ W,
                      const float* __restrict__ bias, float* __restrict__ Cf,
                      _Float16* __restrict__ Ch, int K) {
  __shared__ _Float16 Bs[BROWS * LDS_STRIDE];            // 33 KB
  const int lane = threadIdx.x & 31;
  const int wave = threadIdx.x >> 5;
  const int row0  = (blockIdx.y << 7) + (wave << 5);     // 32-row band per wave
  const int n0blk = blockIdx.x << 6;                     // 64 cols per block
  v8f c00 = {}, c01 = {}, c02 = {}, c03 = {};
  v8f c10 = {}, c11 = {}, c12 = {}, c13 = {};

  for (int kc = 0; kc < K; kc += KCHUNK) {
    __syncthreads();                                     // panel reuse: all readers done
    if (wave == 0) {
      tdm_load_weight_tile(W, K, n0blk, kc, Bs);
      __builtin_amdgcn_s_wait_tensorcnt(0);
    }
    __syncthreads();                                     // publish LDS tile

#pragma unroll 2
    for (int k0 = 0; k0 < KCHUNK; k0 += 32) {
      const int kg = kc + k0;
      if (kg + 128 <= K) {                               // prefetch A stream ahead
        const int m = lane & 15;
        __builtin_prefetch(A + (size_t)(row0 + m) * K + kg + 64, 0, 1);
        __builtin_prefetch(A + (size_t)(row0 + 16 + m) * K + kg + 64, 0, 1);
      }
      v16h a0 = load_fragA(A, (size_t)K, row0,      kg, lane);
      v16h a1 = load_fragA(A, (size_t)K, row0 + 16, kg, lane);
      v16h b0 = load_fragB_lds(Bs, 0,  k0, lane);
      v16h b1 = load_fragB_lds(Bs, 16, k0, lane);
      v16h b2 = load_fragB_lds(Bs, 32, k0, lane);
      v16h b3 = load_fragB_lds(Bs, 48, k0, lane);
      c00 = __builtin_amdgcn_wmma_f32_16x16x32_f16(false, a0, false, b0, (short)0, c00, false, false);
      c01 = __builtin_amdgcn_wmma_f32_16x16x32_f16(false, a0, false, b1, (short)0, c01, false, false);
      c02 = __builtin_amdgcn_wmma_f32_16x16x32_f16(false, a0, false, b2, (short)0, c02, false, false);
      c03 = __builtin_amdgcn_wmma_f32_16x16x32_f16(false, a0, false, b3, (short)0, c03, false, false);
      c10 = __builtin_amdgcn_wmma_f32_16x16x32_f16(false, a1, false, b0, (short)0, c10, false, false);
      c11 = __builtin_amdgcn_wmma_f32_16x16x32_f16(false, a1, false, b1, (short)0, c11, false, false);
      c12 = __builtin_amdgcn_wmma_f32_16x16x32_f16(false, a1, false, b2, (short)0, c12, false, false);
      c13 = __builtin_amdgcn_wmma_f32_16x16x32_f16(false, a1, false, b3, (short)0, c13, false, false);
    }
  }
  const size_t ldc = (size_t)gridDim.x << 6;             // total N
  const int n = lane & 15;
  store_tile<EPI>(c00, Cf, Ch, bias, ldc, row0,      n0blk +      n, lane);
  store_tile<EPI>(c01, Cf, Ch, bias, ldc, row0,      n0blk + 16 + n, lane);
  store_tile<EPI>(c02, Cf, Ch, bias, ldc, row0,      n0blk + 32 + n, lane);
  store_tile<EPI>(c03, Cf, Ch, bias, ldc, row0,      n0blk + 48 + n, lane);
  store_tile<EPI>(c10, Cf, Ch, bias, ldc, row0 + 16, n0blk +      n, lane);
  store_tile<EPI>(c11, Cf, Ch, bias, ldc, row0 + 16, n0blk + 16 + n, lane);
  store_tile<EPI>(c12, Cf, Ch, bias, ldc, row0 + 16, n0blk + 32 + n, lane);
  store_tile<EPI>(c13, Cf, Ch, bias, ldc, row0 + 16, n0blk + 48 + n, lane);
}

// ------------------------------ prep kernels -------------------------------
__global__ __launch_bounds__(256)
void f32_to_f16_kernel(const float* __restrict__ src, _Float16* __restrict__ dst, int n) {
  const int i = blockIdx.x * 256 + threadIdx.x;
  if (i < n) dst[i] = (_Float16)src[i];
}

// Interleave the 4 modality tensors: row r = tok*4 + mod, 512 cols, f16.
__global__ __launch_bounds__(256)
void build_fcat_kernel(const float* __restrict__ x0, const float* __restrict__ x1,
                       const float* __restrict__ x2, const float* __restrict__ x3,
                       _Float16* __restrict__ fcat) {
  const size_t i = (size_t)blockIdx.x * 256 + threadIdx.x;   // < NROWS*512
  const int col = (int)(i & 511);
  const size_t r = i >> 9;
  const int mod = (int)(r & 3);
  const size_t tok = r >> 2;
  const float* x = (mod == 0) ? x0 : (mod == 1) ? x1 : (mod == 2) ? x2 : x3;
  fcat[i] = (_Float16)x[tok * MFEAT + col];
}

// ------------------------------ attention ----------------------------------
// One thread per (token, head): 4x4 scores over dh=64, softmax, 4x64 output.
// qkv rows are (tok*4 + mod), 1536 cols = [q(512) | k(512) | v(512)], f16.
__global__ __launch_bounds__(256)
void attn_kernel(const _Float16* __restrict__ qkv, _Float16* __restrict__ ocat) {
  const int gid = blockIdx.x * 256 + threadIdx.x;    // 0 .. NTOK*NHEADS-1
  const int h = gid & (NHEADS - 1);
  const int n = gid >> 3;
  const _Float16* base = qkv + (size_t)n * (NMODS * 3 * MFEAT);
  const int ho = h * DHEAD;

  float sc[4][4];
#pragma unroll
  for (int s = 0; s < 4; ++s) {
    const _Float16* qp = base + (size_t)s * (3 * MFEAT) + ho;
#pragma unroll
    for (int t = 0; t < 4; ++t) {
      const _Float16* kp = base + (size_t)t * (3 * MFEAT) + MFEAT + ho;
      float acc = 0.f;
#pragma unroll
      for (int dc = 0; dc < DHEAD; dc += 8) {
        v8h qv = *(const v8h*)(qp + dc);
        v8h kv = *(const v8h*)(kp + dc);
#pragma unroll
        for (int j = 0; j < 8; ++j) acc += (float)qv[j] * (float)kv[j];
      }
      sc[s][t] = acc * 0.125f;   // 1/sqrt(64)
    }
  }
#pragma unroll
  for (int s = 0; s < 4; ++s) {
    const float mx = fmaxf(fmaxf(sc[s][0], sc[s][1]), fmaxf(sc[s][2], sc[s][3]));
    float e0 = expf(sc[s][0] - mx), e1 = expf(sc[s][1] - mx);
    float e2 = expf(sc[s][2] - mx), e3 = expf(sc[s][3] - mx);
    const float inv = 1.0f / (e0 + e1 + e2 + e3);
    sc[s][0] = e0 * inv; sc[s][1] = e1 * inv; sc[s][2] = e2 * inv; sc[s][3] = e3 * inv;
  }
#pragma unroll
  for (int s = 0; s < 4; ++s) {
    _Float16* op = ocat + ((size_t)n * 4 + s) * MFEAT + ho;
#pragma unroll
    for (int dc = 0; dc < DHEAD; dc += 8) {
      float ov[8] = {0.f, 0.f, 0.f, 0.f, 0.f, 0.f, 0.f, 0.f};
#pragma unroll
      for (int t = 0; t < 4; ++t) {
        v8h vv = *(const v8h*)(base + (size_t)t * (3 * MFEAT) + 2 * MFEAT + ho + dc);
#pragma unroll
        for (int j = 0; j < 8; ++j) ov[j] += sc[s][t] * (float)vv[j];
      }
      v8h oh;
#pragma unroll
      for (int j = 0; j < 8; ++j) oh[j] = (_Float16)ov[j];
      *(v8h*)(op + dc) = oh;
    }
  }
}

// ------------------------------ layernorms ---------------------------------
__device__ __forceinline__ float wave_sum32(float v) {
#pragma unroll
  for (int m = 16; m >= 1; m >>= 1) v += __shfl_xor(v, m, 32);
  return v;
}

// One wave per token: h1 = LN(F + attn_out) per modality row, agg = sum_s h1.
__global__ __launch_bounds__(256)
void ln1_agg_kernel(const float* __restrict__ x0, const float* __restrict__ x1,
                    const float* __restrict__ x2, const float* __restrict__ x3,
                    const float* __restrict__ attn_out,
                    const float* __restrict__ g, const float* __restrict__ b,
                    float* __restrict__ agg, _Float16* __restrict__ agg16) {
  const int wave = threadIdx.x >> 5;
  const int lane = threadIdx.x & 31;
  const size_t tok = (size_t)blockIdx.x * 8 + wave;
  const float* xs[4] = {x0, x1, x2, x3};
  float acc[16];
#pragma unroll
  for (int j = 0; j < 16; ++j) acc[j] = 0.f;
#pragma unroll
  for (int s = 0; s < 4; ++s) {
    const size_t rbase = (tok * 4 + s) * MFEAT;
    float t[16];
    float sum = 0.f;
#pragma unroll
    for (int j = 0; j < 16; ++j) {
      const int col = lane + 32 * j;
      t[j] = xs[s][tok * MFEAT + col] + attn_out[rbase + col];
      sum += t[j];
    }
    const float mu = wave_sum32(sum) * (1.0f / MFEAT);
    float var = 0.f;
#pragma unroll
    for (int j = 0; j < 16; ++j) { const float d = t[j] - mu; var += d * d; }
    var = wave_sum32(var) * (1.0f / MFEAT);
    const float rs = rsqrtf(var + 1e-5f);
#pragma unroll
    for (int j = 0; j < 16; ++j) {
      const int col = lane + 32 * j;
      acc[j] += (t[j] - mu) * rs * g[col] + b[col];
    }
  }
#pragma unroll
  for (int j = 0; j < 16; ++j) {
    const int col = lane + 32 * j;
    agg[tok * MFEAT + col]   = acc[j];
    agg16[tok * MFEAT + col] = (_Float16)acc[j];
  }
}

// One wave per token: out = LN(agg + ffn2).
__global__ __launch_bounds__(256)
void ln2_kernel(const float* __restrict__ agg, const float* __restrict__ ffn2,
                const float* __restrict__ g, const float* __restrict__ b,
                float* __restrict__ out) {
  const int wave = threadIdx.x >> 5;
  const int lane = threadIdx.x & 31;
  const size_t tok = (size_t)blockIdx.x * 8 + wave;
  float t[16];
  float sum = 0.f;
#pragma unroll
  for (int j = 0; j < 16; ++j) {
    const int col = lane + 32 * j;
    t[j] = agg[tok * MFEAT + col] + ffn2[tok * MFEAT + col];
    sum += t[j];
  }
  const float mu = wave_sum32(sum) * (1.0f / MFEAT);
  float var = 0.f;
#pragma unroll
  for (int j = 0; j < 16; ++j) { const float d = t[j] - mu; var += d * d; }
  var = wave_sum32(var) * (1.0f / MFEAT);
  const float rs = rsqrtf(var + 1e-5f);
#pragma unroll
  for (int j = 0; j < 16; ++j) {
    const int col = lane + 32 * j;
    out[tok * MFEAT + col] = (t[j] - mu) * rs * g[col] + b[col];
  }
}

// ------------------------------ launcher -----------------------------------
extern "C" void kernel_launch(void* const* d_in, const int* in_sizes, int n_in,
                              void* d_out, int out_size, void* d_ws, size_t ws_size,
                              hipStream_t stream) {
  (void)in_sizes; (void)n_in; (void)out_size; (void)ws_size;
  const float* x0    = (const float*)d_in[0];
  const float* x1    = (const float*)d_in[1];
  const float* x2    = (const float*)d_in[2];
  const float* x3    = (const float*)d_in[3];
  const float* in_w  = (const float*)d_in[4];   // (1536, 512)
  const float* in_b  = (const float*)d_in[5];   // (1536,)
  const float* out_w = (const float*)d_in[6];   // (512, 512)
  const float* out_b = (const float*)d_in[7];   // (512,)
  const float* ln1g  = (const float*)d_in[8];
  const float* ln1b  = (const float*)d_in[9];
  const float* w1    = (const float*)d_in[10];  // (2048, 512)
  const float* b1    = (const float*)d_in[11];
  const float* w2    = (const float*)d_in[12];  // (512, 2048)
  const float* b2    = (const float*)d_in[13];
  const float* ln2g  = (const float*)d_in[14];
  const float* ln2b  = (const float*)d_in[15];

  // ---- workspace layout (lifetime-aliased regions) ----
  char* ws = (char*)d_ws;
  const size_t o_winh  = 0;                               // 1536*512*2
  const size_t o_woh   = o_winh + (size_t)1536 * 512 * 2; //  512*512*2
  const size_t o_w1h   = o_woh  + (size_t)512 * 512 * 2;  // 2048*512*2
  const size_t o_w2h   = o_w1h  + (size_t)2048 * 512 * 2; //  512*2048*2
  const size_t o_reg1  = o_w2h  + (size_t)512 * 2048 * 2; // fcat / ocat / ffn2
  const size_t o_reg2  = o_reg1 + (size_t)NROWS * MFEAT * 2;      // qkv / attn_out / ffn1
  const size_t o_agg   = o_reg2 + (size_t)NROWS * 3 * MFEAT * 2;  // agg f32
  const size_t o_agg16 = o_agg  + (size_t)NTOK * MFEAT * 4;       // agg f16

  _Float16* winh  = (_Float16*)(ws + o_winh);
  _Float16* woh   = (_Float16*)(ws + o_woh);
  _Float16* w1h   = (_Float16*)(ws + o_w1h);
  _Float16* w2h   = (_Float16*)(ws + o_w2h);
  _Float16* fcat  = (_Float16*)(ws + o_reg1);   // (65536, 512) f16
  _Float16* ocat  = (_Float16*)(ws + o_reg1);   // reuse after fcat dead
  float*    ffn2  = (float*)   (ws + o_reg1);   // reuse after ocat dead
  _Float16* qkvh  = (_Float16*)(ws + o_reg2);   // (65536, 1536) f16
  float*    attno = (float*)   (ws + o_reg2);   // reuse after qkv dead
  _Float16* ffn1h = (_Float16*)(ws + o_reg2);   // reuse after attn_out dead
  float*    agg   = (float*)   (ws + o_agg);
  _Float16* agg16 = (_Float16*)(ws + o_agg16);
  float*    outp  = (float*)d_out;

  // ---- 0) weight conversion + activation gather ----
  f32_to_f16_kernel<<<(1536 * 512) / 256, 256, 0, stream>>>(in_w,  winh, 1536 * 512);
  f32_to_f16_kernel<<<(512  * 512) / 256, 256, 0, stream>>>(out_w, woh,  512  * 512);
  f32_to_f16_kernel<<<(2048 * 512) / 256, 256, 0, stream>>>(w1,    w1h,  2048 * 512);
  f32_to_f16_kernel<<<(512 * 2048) / 256, 256, 0, stream>>>(w2,    w2h,  512 * 2048);
  build_fcat_kernel<<<(int)(((size_t)NROWS * MFEAT) / 256), 256, 0, stream>>>(x0, x1, x2, x3, fcat);

  // ---- 1) QKV GEMM: (65536,512) x (512,1536) -> f16 ----
  gemm_wmma_kernel<1><<<dim3(1536 / 64, NROWS / 128), 128, 0, stream>>>(
      fcat, winh, in_b, nullptr, qkvh, MFEAT);

  // ---- 2) per-(token,head) 4x4 attention ----
  attn_kernel<<<(NTOK * NHEADS) / 256, 256, 0, stream>>>(qkvh, ocat);

  // ---- 3) out-proj GEMM: (65536,512) x (512,512) -> f32 ----
  gemm_wmma_kernel<0><<<dim3(512 / 64, NROWS / 128), 128, 0, stream>>>(
      ocat, woh, out_b, attno, nullptr, MFEAT);

  // ---- 4) LN1 + residual + modality aggregation ----
  ln1_agg_kernel<<<NTOK / 8, 256, 0, stream>>>(x0, x1, x2, x3, attno, ln1g, ln1b, agg, agg16);

  // ---- 5) FFN1 GEMM + GELU: (16384,512) x (512,2048) -> f16 ----
  gemm_wmma_kernel<2><<<dim3(2048 / 64, NTOK / 128), 128, 0, stream>>>(
      agg16, w1h, b1, nullptr, ffn1h, MFEAT);

  // ---- 6) FFN2 GEMM: (16384,2048) x (2048,512) -> f32 ----
  gemm_wmma_kernel<0><<<dim3(512 / 64, NTOK / 128), 128, 0, stream>>>(
      ffn1h, w2h, b2, ffn2, nullptr, FFNDIM);

  // ---- 7) LN2 -> output ----
  ln2_kernel<<<NTOK / 8, 256, 0, stream>>>(agg, ffn2, ln2g, ln2b, outp);
}
